// LSTM_4793183502526
// MI455X (gfx1250) — compile-verified, matrix operand order checked by reading, not run on hardware
//
#include <hip/hip_runtime.h>
#include <stdint.h>

// ---------------------------------------------------------------------------
// MI455X (gfx1250) LSTM stack: 6 layers, B=32, T=1024, H=272.
//
// Plan (see analysis):
//  * per layer: hoisted input GEMM xW = h_seq @ W_ih^T + (b_ih+b_hh)
//      -> parallel WMMA bf16 GEMM over (B*T_chunk) x 4H x Hpad
//  * recurrence: persistent single-workgroup scan kernel, 17 waves (one per
//      16-col h tile), h ping-pong in LDS (36KB), c + mean-pool in registers,
//      W_hh bf16 streamed from L2 (resident: 3.7MB), 1224 WMMA / step.
//  * head: feat = sum_h / T, scores = feat @ W_r^T + b_r.
// All K dims padded 272 -> 288 (zeros) for clean 9 x K32 WMMA tiles.
// ---------------------------------------------------------------------------

typedef __attribute__((ext_vector_type(16))) __bf16 bf16x16;
typedef __attribute__((ext_vector_type(8)))  float  f32x8;

union Frag { bf16x16 v; uint4 q[2]; };

#define Bz 32
#define Tt 1024
#define Hh 272
#define Hp 288        // padded hidden (K) size
#define G4 1088       // 4*H gate dim
#define NL 6
#define TC 128        // time chunk for xW staging

__device__ __forceinline__ unsigned short f2bf(float f) {
  uint32_t u = __float_as_uint(f);
  u += 0x7FFFu + ((u >> 16) & 1u);   // round-to-nearest-even
  return (unsigned short)(u >> 16);
}
__device__ __forceinline__ float sigm(float x) { return 1.0f / (1.0f + __expf(-x)); }

__device__ __forceinline__ f32x8 wmma_bf16(const Frag& a, const Frag& b, f32x8 c) {
  // D = A(16x32 bf16) x B(32x16 bf16) + C(16x16 f32)
  return __builtin_amdgcn_wmma_f32_16x16x32_bf16(
      /*neg_a=*/false, a.v, /*neg_b=*/false, b.v,
      /*c_mod=*/(short)0, c, /*reuse_a=*/false, /*reuse_b=*/false);
}

// f32 [rows][272] -> bf16 [rows][288] with zero K-pad (used for x, W_ih, W_hh)
__global__ void k_cvt_pad(const float* __restrict__ in, unsigned short* __restrict__ out, int n) {
  int i = blockIdx.x * blockDim.x + threadIdx.x;
  if (i >= n) return;
  int j = i % Hp, r = i / Hp;
  out[i] = (j < Hh) ? f2bf(in[r * Hh + j]) : (unsigned short)0;
}

// zero the K-pad columns of a bf16 [rows][Hp] sequence buffer
__global__ void k_zero_pad(unsigned short* __restrict__ o, int n) {
  int i = blockIdx.x * blockDim.x + threadIdx.x;
  if (i >= n) return;
  o[(i / 16) * Hp + Hh + (i % 16)] = 0;
}

__global__ void k_zero_f32(float* __restrict__ o, int n) {
  int i = blockIdx.x * blockDim.x + threadIdx.x;
  if (i < n) o[i] = 0.0f;
}

// ---------------------------------------------------------------------------
// Input GEMM: xW[mrow][g] = sum_k hseq[row][k] * W[g][k] + b_ih[g] + b_hh[g]
// mrow = b*TC + tl (chunk-local), A row = b*Tt + t0 + tl.
// Wave tile: 32(M) x 32(N), K = 9 x 32. Grid = 544 blocks x 8 waves = 128x34.
// ---------------------------------------------------------------------------
__global__ void __launch_bounds__(256) k_gemm(
    const unsigned short* __restrict__ A,   // hseq in, [B*T][Hp] bf16
    const unsigned short* __restrict__ W,   // W_ih layer, [G4][Hp] bf16
    const float* __restrict__ bi, const float* __restrict__ bh,
    float* __restrict__ xW, int t0)
{
  int lane = threadIdx.x & 31;
  int wave = blockIdx.x * (blockDim.x >> 5) + (threadIdx.x >> 5);
  int mp = wave & 127;      // 128 M-pairs (M tiles of 32 over 4096 rows)
  int np = wave >> 7;       // 34 N-pairs  (N tiles of 32 over 1088 cols)
  int m = lane & 15, half = lane >> 4;

  f32x8 acc[2][2] = {};
  int arow[2], grow[2];
#pragma unroll
  for (int mt = 0; mt < 2; ++mt) {
    int rloc = mp * 32 + mt * 16 + m;                 // chunk-local (b,tl) row
    arow[mt] = (rloc >> 7) * Tt + t0 + (rloc & 127);  // global seq row
  }
#pragma unroll
  for (int nt = 0; nt < 2; ++nt) grow[nt] = np * 32 + nt * 16 + m;

#pragma unroll
  for (int kt = 0; kt < 9; ++kt) {
    int k0 = kt * 32 + half * 8;
    Frag a[2], b[2];
#pragma unroll
    for (int mt = 0; mt < 2; ++mt) {
      const uint4* p = (const uint4*)(A + (size_t)arow[mt] * Hp + k0);
      a[mt].q[0] = p[0]; a[mt].q[1] = p[2];           // +16 elems = +32B
    }
#pragma unroll
    for (int nt = 0; nt < 2; ++nt) {
      const uint4* p = (const uint4*)(W + (size_t)grow[nt] * Hp + k0);
      b[nt].q[0] = p[0]; b[nt].q[1] = p[2];
    }
#pragma unroll
    for (int mt = 0; mt < 2; ++mt)
#pragma unroll
      for (int nt = 0; nt < 2; ++nt)
        acc[mt][nt] = wmma_bf16(a[mt], b[nt], acc[mt][nt]);
  }

#pragma unroll
  for (int nt = 0; nt < 2; ++nt) {
    int g = np * 32 + nt * 16 + m;
    float bias = bi[g] + bh[g];
#pragma unroll
    for (int mt = 0; mt < 2; ++mt)
#pragma unroll
      for (int r = 0; r < 8; ++r) {
        int mrow = mp * 32 + mt * 16 + r + half * 8;
        xW[(size_t)mrow * G4 + g] = acc[mt][nt][r] + bias;
      }
  }
}

// ---------------------------------------------------------------------------
// Recurrent scan: one workgroup, 17 waves. Wave w owns h columns [16w,16w+16).
// Per step: gates = xW_t + h @ W_hh^T (2 M-tiles x 4 gate-tiles x 9 K WMMAs),
// lane-local LSTM elementwise, h -> LDS (ping-pong) + global seq out.
// ---------------------------------------------------------------------------
__global__ void __launch_bounds__(544) k_scan(
    const float* __restrict__ xW,            // [B*TC][G4] f32 gate pre-acts
    const unsigned short* __restrict__ W,    // W_hh layer, [G4][Hp] bf16
    unsigned short* __restrict__ hout,       // layer output seq [B*T][Hp] bf16
    unsigned short* __restrict__ hstate,     // [B][Hp] bf16 carry
    float* __restrict__ cstate,              // [B][Hh] f32 carry
    float* __restrict__ featsum,             // [B][Hh] f32 mean-pool accum
    int t0, int first, int accf)
{
  __shared__ unsigned short hls[2][Bz * Hp]; // 36,864 B of 320KB WGP LDS

  int tid = threadIdx.x, w = tid >> 5, lane = tid & 31;
  int n = lane & 15, half = lane >> 4;

  for (int i = tid; i < Bz * Hp; i += 544) {
    int j = i % Hp;
    hls[0][i] = (!first && j < Hh) ? hstate[i] : (unsigned short)0;
    hls[1][i] = 0;                             // keep pads zero
  }

  f32x8 cc[2], fa[2];
#pragma unroll
  for (int mt = 0; mt < 2; ++mt)
#pragma unroll
    for (int r = 0; r < 8; ++r) {
      int b = mt * 16 + r + half * 8;
      int j = w * 16 + n;
      cc[mt][r] = first ? 0.0f : cstate[b * Hh + j];
      fa[mt][r] = 0.0f;
    }

  int grow[4];                                 // PyTorch gate order i,f,g,o
#pragma unroll
  for (int gi = 0; gi < 4; ++gi) grow[gi] = gi * Hh + w * 16 + n;

  __syncthreads();

  for (int tl = 0; tl < TC; ++tl) {
    int cur = tl & 1, nxt = cur ^ 1;
    f32x8 acc[2][4];
#pragma unroll
    for (int mt = 0; mt < 2; ++mt)
#pragma unroll
      for (int gi = 0; gi < 4; ++gi)
#pragma unroll
        for (int r = 0; r < 8; ++r) {
          int b = mt * 16 + r + half * 8;
          acc[mt][gi][r] = xW[(size_t)(b * TC + tl) * G4 + grow[gi]];
        }

#pragma unroll
    for (int kt = 0; kt < 9; ++kt) {
      int k0 = kt * 32 + half * 8;
      Frag a[2];
#pragma unroll
      for (int mt = 0; mt < 2; ++mt) {
        const uint4* p = (const uint4*)&hls[cur][(mt * 16 + (lane & 15)) * Hp + k0];
        a[mt].q[0] = p[0]; a[mt].q[1] = p[2];
      }
#pragma unroll
      for (int gi = 0; gi < 4; ++gi) {
        Frag b;
        const uint4* p = (const uint4*)(W + (size_t)grow[gi] * Hp + k0);
        b.q[0] = p[0]; b.q[1] = p[2];
        acc[0][gi] = wmma_bf16(a[0], b, acc[0][gi]);
        acc[1][gi] = wmma_bf16(a[1], b, acc[1][gi]);
      }
    }

#pragma unroll
    for (int mt = 0; mt < 2; ++mt)
#pragma unroll
      for (int r = 0; r < 8; ++r) {
        float iv = sigm(acc[mt][0][r]);
        float fv = sigm(acc[mt][1][r]);
        float gv = tanhf(acc[mt][2][r]);
        float ov = sigm(acc[mt][3][r]);
        float c  = fv * cc[mt][r] + iv * gv;
        cc[mt][r] = c;
        float h  = ov * tanhf(c);
        fa[mt][r] += h;
        unsigned short hb = f2bf(h);
        int b = mt * 16 + r + half * 8;
        int j = w * 16 + n;
        hls[nxt][b * Hp + j] = hb;
        hout[(size_t)(b * Tt + t0 + tl) * Hp + j] = hb;
      }
    __syncthreads();
  }

  // carry state out (final h lives in buffer 0 since TC is even)
#pragma unroll
  for (int mt = 0; mt < 2; ++mt)
#pragma unroll
    for (int r = 0; r < 8; ++r) {
      int b = mt * 16 + r + half * 8;
      int j = w * 16 + n;
      cstate[b * Hh + j] = cc[mt][r];
      hstate[b * Hp + j] = hls[0][b * Hp + j];
      if (accf) featsum[b * Hh + j] += fa[mt][r];
    }
}

// scores[b][s] = (featsum[b]/T) . W_r[s] + b_r[s]; feat out = featsum/T
__global__ void k_head(const float* __restrict__ featsum, const float* __restrict__ Wr,
                       const float* __restrict__ br, float* __restrict__ out)
{
  int i = blockIdx.x * blockDim.x + threadIdx.x;
  const float inv = 1.0f / (float)Tt;
  if (i < Bz * 3) {
    int b = i / 3, s = i % 3;
    float acc = br[s];
    for (int j = 0; j < Hh; ++j) acc += featsum[b * Hh + j] * inv * Wr[s * Hh + j];
    out[i] = acc;
  }
  int f = i - Bz * 3;
  if (f >= 0 && f < Bz * Hh) out[Bz * 3 + f] = featsum[f] * inv;
}

extern "C" void kernel_launch(void* const* d_in, const int* in_sizes, int n_in,
                              void* d_out, int out_size, void* d_ws, size_t ws_size,
                              hipStream_t stream)
{
  const float* x   = (const float*)d_in[0];
  const float* Wih = (const float*)d_in[1];
  const float* Whh = (const float*)d_in[2];
  const float* bih = (const float*)d_in[3];
  const float* bhh = (const float*)d_in[4];
  const float* Wr  = (const float*)d_in[5];
  const float* br  = (const float*)d_in[6];

  char* ws = (char*)d_ws;
  size_t oWih = 0;
  size_t oWhh = oWih + (size_t)NL * G4 * Hp * 2;    //  3.76 MB
  size_t oA   = oWhh + (size_t)NL * G4 * Hp * 2;    //  3.76 MB
  size_t oB   = oA   + (size_t)Bz * Tt * Hp * 2;    // 18.87 MB
  size_t oXW  = oB   + (size_t)Bz * Tt * Hp * 2;    // 18.87 MB
  size_t oHS  = oXW  + (size_t)Bz * TC * G4 * 4;    // 17.83 MB
  size_t oCS  = oHS  + (size_t)Bz * Hp * 2;
  size_t oFS  = oCS  + (size_t)Bz * Hh * 4;         // total ~60.3 MB

  unsigned short* WihB  = (unsigned short*)(ws + oWih);
  unsigned short* WhhB  = (unsigned short*)(ws + oWhh);
  unsigned short* hseqA = (unsigned short*)(ws + oA);
  unsigned short* hseqB = (unsigned short*)(ws + oB);
  float*          xWb   = (float*)(ws + oXW);
  unsigned short* hstate= (unsigned short*)(ws + oHS);
  float*          cstate= (float*)(ws + oCS);
  float*          fsum  = (float*)(ws + oFS);

  {
    int nx = Bz * Tt * Hp;
    k_cvt_pad<<<(nx + 255) / 256, 256, 0, stream>>>(x, hseqA, nx);
    int nw = NL * G4 * Hp;
    k_cvt_pad<<<(nw + 255) / 256, 256, 0, stream>>>(Wih, WihB, nw);
    k_cvt_pad<<<(nw + 255) / 256, 256, 0, stream>>>(Whh, WhhB, nw);
    int npd = Bz * Tt * 16;
    k_zero_pad<<<(npd + 255) / 256, 256, 0, stream>>>(hseqB, npd);
    k_zero_f32<<<(Bz * Hh + 255) / 256, 256, 0, stream>>>(fsum, Bz * Hh);
  }

  unsigned short* hin  = hseqA;
  unsigned short* hout = hseqB;
  for (int l = 0; l < NL; ++l) {
    const unsigned short* Wl = WihB + (size_t)l * G4 * Hp;
    const unsigned short* Wh = WhhB + (size_t)l * G4 * Hp;
    const float* bil = bih + (size_t)l * G4;
    const float* bhl = bhh + (size_t)l * G4;
    for (int ch = 0; ch < Tt / TC; ++ch) {
      int t0 = ch * TC;
      k_gemm<<<544, 256, 0, stream>>>(hin, Wl, bil, bhl, xWb, t0);
      k_scan<<<1, 544, 0, stream>>>(xWb, Wh, hout, hstate, cstate, fsum,
                                    t0, ch == 0 ? 1 : 0, (l == NL - 1) ? 1 : 0);
    }
    unsigned short* t = hin; hin = hout; hout = t;
  }

  k_head<<<35, 256, 0, stream>>>(fsum, Wr, br, (float*)d_out);
}
